// LSTMAttentionPooling_89232240542330
// MI455X (gfx1250) — matrix operand order repaired and checked by reading, not compiled, not run on hardware
//
#include <hip/hip_runtime.h>
#include <hip/hip_bf16.h>
#include <stdint.h>

// ---------------- CDNA5 WMMA types ----------------
typedef __attribute__((ext_vector_type(16))) __bf16 v16bf;
typedef __attribute__((ext_vector_type(8)))  __bf16 v8bf;
typedef __attribute__((ext_vector_type(8)))  float  v8f;

// ---------------- problem constants ----------------
#define NGRAPH 8192
#define SEQL   64
#define DIM    128          // input dim == hidden dim
#define GATES  512          // 4*H
#define KTOT   256          // K of fused GEMM: [x ; h]
#define GT     32           // graphs (M rows) per block = 2 WMMA M-tiles
#define NBLK   (NGRAPH/GT)  // 256 blocks (1 per WGP, LDS-limited)
#define NTHR   256          // 8 waves of 32
#define NWAVE  8

// ---------------- LDS layout (ushort units) ----------------
#define W_OFF   0
#define W_CNT   (GATES*KTOT)            // 131072 ushorts = 256 KB packed weights
#define X_OFF   (W_OFF + W_CNT)         // x tile, double buffered
#define X_CNT   (2*GT*DIM)              // 8192 ushorts = 16 KB
#define H_OFF   (X_OFF + X_CNT)         // h_{t-1} tile (bf16)
#define H_CNT   (GT*DIM)                // 4096 ushorts = 8 KB
#define S_OFF   (H_OFF + H_CNT)         // 64 floats of scores (double buffered)
#define SMEM_USHORT (S_OFF + 128)
#define SMEM_BYTES  (SMEM_USHORT * 2)   // ~280.3 KB < 320 KB/WGP

// ---------------- helpers ----------------
__device__ __forceinline__ unsigned f2bf_u(float f) {
  unsigned u = __float_as_uint(f);
  return (u + 0x7FFFu + ((u >> 16) & 1u)) >> 16;   // round-to-nearest-even bf16
}
__device__ __forceinline__ unsigned short f2bf(float f) { return (unsigned short)f2bf_u(f); }

__device__ __forceinline__ uint4 packbf8(float4 a, float4 b) {
  uint4 r;
  r.x = f2bf_u(a.x) | (f2bf_u(a.y) << 16);
  r.y = f2bf_u(a.z) | (f2bf_u(a.w) << 16);
  r.z = f2bf_u(b.x) | (f2bf_u(b.y) << 16);
  r.w = f2bf_u(b.z) | (f2bf_u(b.w) << 16);
  return r;
}

// two contiguous 16-byte LDS reads -> one 32-byte WMMA fragment
__device__ __forceinline__ v16bf ld_frag(const unsigned short* p0, const unsigned short* p1) {
  v8bf lo = *(const v8bf*)p0;
  v8bf hi = *(const v8bf*)p1;
  return __builtin_shufflevector(lo, hi, 0,1,2,3,4,5,6,7,8,9,10,11,12,13,14,15);
}

__device__ __forceinline__ float sigm(float x) {
  return __builtin_amdgcn_rcpf(1.0f + __expf(-x));
}
__device__ __forceinline__ float tanhfast(float x) {
  return 1.0f - 2.0f * __builtin_amdgcn_rcpf(__expf(2.0f * x) + 1.0f);
}

// ---------------- prep: pack [W_ih | W_hh] -> bf16 [512][256], fold biases ----
__global__ void lstm_prep_kernel(const float* __restrict__ wih,
                                 const float* __restrict__ whh,
                                 const float* __restrict__ bih,
                                 const float* __restrict__ bhh,
                                 unsigned short* __restrict__ wpack,
                                 float* __restrict__ bias) {
  int idx = blockIdx.x * blockDim.x + threadIdx.x;   // exactly 512*256 = 131072
  int n = idx >> 8;          // gate column 0..511
  int k = idx & 255;         // fused K 0..255
  float v = (k < DIM) ? wih[n * DIM + k] : whh[n * DIM + (k - DIM)];
  wpack[n * KTOT + k] = f2bf(v);
  if (idx < GATES) bias[idx] = bih[idx] + bhh[idx];
}

// ---------------- fused LSTM + online-softmax attention pooling ----------------
__global__ __launch_bounds__(NTHR, 1)
void lstm_attn_kernel(const float* __restrict__ x,
                      const unsigned short* __restrict__ wpack,
                      const float* __restrict__ bias,
                      const float* __restrict__ watt,
                      float* __restrict__ out) {
  extern __shared__ unsigned short sm[];
  const int tid  = threadIdx.x;
  const int lane = tid & 31;
  const int wv   = tid >> 5;        // wave id 0..7
  const int g0   = blockIdx.x * GT; // first graph of this tile
  const int mlo  = lane & 15;
  const int hi16 = lane >> 4;       // 0: rows 0..7 / K-lo ; 1: rows 8..15 / K-hi

  // --- stage packed weights into LDS with CDNA5 async copies (ASYNCcnt) ---
  {
    unsigned ldsbase = (unsigned)(uintptr_t)(void*)(sm + W_OFF); // low 32b = DS offset
    uint64_t gbase   = (uint64_t)(uintptr_t)wpack;
    #pragma unroll 4
    for (int i = 0; i < (W_CNT * 2 / 16) / NTHR; ++i) {          // 64 x 16B per thread
      unsigned off = (unsigned)(tid + i * NTHR) * 16u;
      unsigned la  = ldsbase + off;
      uint64_t ga  = gbase + off;
      asm volatile("global_load_async_to_lds_b128 %0, %1, off"
                   :: "v"(la), "v"(ga) : "memory");
    }
    asm volatile("s_wait_asynccnt 0x0" ::: "memory");
  }

  // --- zero h_{-1} and score buffers ---
  for (int i = tid; i < H_CNT; i += NTHR) sm[H_OFF + i] = 0;
  float* scoreF = (float*)(sm + S_OFF);
  if (tid < 64) scoreF[tid] = 0.0f;

  // --- per-thread x staging role: (graph row, 16-wide column chunk) ---
  const int xi = tid >> 3;          // 0..31: graph row in tile
  const int xc = (tid & 7) * 16;    // column base
  const float* xrow = x + ((size_t)(g0 + xi) * SEQL) * DIM + xc;
  {
    float4 f0 = *(const float4*)(xrow + 0);
    float4 f1 = *(const float4*)(xrow + 4);
    float4 f2 = *(const float4*)(xrow + 8);
    float4 f3 = *(const float4*)(xrow + 12);
    *(uint4*)(sm + X_OFF + xi * DIM + xc)     = packbf8(f0, f1);
    *(uint4*)(sm + X_OFF + xi * DIM + xc + 8) = packbf8(f2, f3);
  }

  // --- per-lane constants: this wave owns hidden units [16*wv, 16*wv+16) ---
  const int u0 = wv * 16 + mlo;            // N column (hidden unit)
  const float wa = watt[u0];               // attention weight for my unit
  float b4[4];
  #pragma unroll
  for (int g = 0; g < 4; ++g) b4[g] = bias[g * DIM + u0];

  // --- register state, C/D layout: tile tl, VGPR r -> row m = tl*16 + r + 8*hi16
  float cst[16], accP[16], Mrun[16], Zrun[16];
  #pragma unroll
  for (int i = 0; i < 16; ++i) { cst[i] = 0.0f; accP[i] = 0.0f; Zrun[i] = 0.0f; Mrun[i] = -1e30f; }

  __syncthreads();

  for (int t = 0; t < SEQL; ++t) {
    const int par = t & 1;

    // prefetch x_{t+1} -> registers; pull x_{t+4} toward WGP$ (global_prefetch_b8)
    uint4 xpk0, xpk1;
    const bool havex = (t + 1 < SEQL);
    if (havex) {
      float4 f0 = *(const float4*)(xrow + (t + 1) * DIM + 0);
      float4 f1 = *(const float4*)(xrow + (t + 1) * DIM + 4);
      float4 f2 = *(const float4*)(xrow + (t + 1) * DIM + 8);
      float4 f3 = *(const float4*)(xrow + (t + 1) * DIM + 12);
      xpk0 = packbf8(f0, f1);
      xpk1 = packbf8(f2, f3);
    }
    if (t + 4 < SEQL) __builtin_prefetch(xrow + (t + 4) * DIM, 0, 3);

    // --- fused GEMM: [32, 256] @ [256, 512] with B-fragment reuse across M ---
    v8f acc[4][2];
    #pragma unroll
    for (int g = 0; g < 4; ++g) {
      float b = b4[g];
      v8f a = {b, b, b, b, b, b, b, b};
      acc[g][0] = a; acc[g][1] = a;
    }

    const unsigned short* xb = sm + X_OFF + par * (GT * DIM);
    const unsigned short* hb = sm + H_OFF;
    #pragma unroll
    for (int kc = 0; kc < 8; ++kc) {
      // A 16x32 bf16: lane<16 K = {0..7,16..23}; lane>=16 K = {8..15,24..31}
      const unsigned short* ab = (kc < 4) ? xb : hb;
      const int o0 = mlo * DIM + (kc & 3) * 32 + hi16 * 8;
      v16bf A0 = ld_frag(ab + o0,            ab + o0 + 16);            // graphs 0..15
      v16bf A1 = ld_frag(ab + o0 + 16 * DIM, ab + o0 + 16 * DIM + 16); // graphs 16..31
      #pragma unroll
      for (int g = 0; g < 4; ++g) {
        // B 32x16: lane n = gate col; lanes 0-15 K {0..7,16..23}, lanes 16-31 K {8..15,24..31}
        const int bo = (g * DIM + u0) * KTOT + kc * 32 + hi16 * 8;
        v16bf B = ld_frag(sm + W_OFF + bo, sm + W_OFF + bo + 16);
        acc[g][0] = __builtin_amdgcn_wmma_f32_16x16x32_bf16(
                        false, A0, false, B, (short)0, acc[g][0], false, false);
        acc[g][1] = __builtin_amdgcn_wmma_f32_16x16x32_bf16(
                        false, A1, false, B, (short)0, acc[g][1], false, false);
      }
    }

    // --- pointwise LSTM cell update, all in registers ---
    float hv[16];
    #pragma unroll
    for (int tl = 0; tl < 2; ++tl) {
      #pragma unroll
      for (int r = 0; r < 8; ++r) {
        const int j = tl * 8 + r;
        float ig = sigm(acc[0][tl][r]);
        float fg = sigm(acc[1][tl][r]);
        float gg = tanhfast(acc[2][tl][r]);
        float og = sigm(acc[3][tl][r]);
        float c  = fg * cst[j] + ig * gg;
        cst[j]   = c;
        hv[j]    = og * tanhfast(c);
      }
    }

    // --- attention score partials: reduce this wave's 16-unit slice ---
    #pragma unroll
    for (int tl = 0; tl < 2; ++tl) {
      #pragma unroll
      for (int r = 0; r < 8; ++r) {
        float s = hv[tl * 8 + r] * wa;
        s += __shfl_xor(s, 1, 16);
        s += __shfl_xor(s, 2, 16);
        s += __shfl_xor(s, 4, 16);
        s += __shfl_xor(s, 8, 16);
        if (mlo == 0)
          atomicAdd(&scoreF[par * 32 + tl * 16 + hi16 * 8 + r], s); // ds_add_f32
      }
    }

    __syncthreads();   // A: all LDS reads + score atomics of step t complete

    // --- online softmax update (b_att cancels in softmax) ---
    #pragma unroll
    for (int tl = 0; tl < 2; ++tl) {
      #pragma unroll
      for (int r = 0; r < 8; ++r) {
        const int j = tl * 8 + r;
        float s  = scoreF[par * 32 + tl * 16 + hi16 * 8 + r];
        float nm = fmaxf(Mrun[j], s);
        float al = __expf(Mrun[j] - nm);
        float p  = __expf(s - nm);
        accP[j]  = accP[j] * al + p * hv[j];
        Zrun[j]  = Zrun[j] * al + p;
        Mrun[j]  = nm;
      }
    }

    // --- publish h_t (bf16), stage x_{t+1}, reset next score buffer ---
    #pragma unroll
    for (int tl = 0; tl < 2; ++tl)
      #pragma unroll
      for (int r = 0; r < 8; ++r)
        sm[H_OFF + (tl * 16 + hi16 * 8 + r) * DIM + u0] = f2bf(hv[tl * 8 + r]);
    if (havex) {
      *(uint4*)(sm + X_OFF + (1 - par) * (GT * DIM) + xi * DIM + xc)     = xpk0;
      *(uint4*)(sm + X_OFF + (1 - par) * (GT * DIM) + xi * DIM + xc + 8) = xpk1;
    }
    if (tid < 32) scoreF[(1 - par) * 32 + tid] = 0.0f;

    __syncthreads();   // B: writes visible before next iteration's reads
  }

  // --- normalized pooled output: [8192, 128] f32 ---
  #pragma unroll
  for (int tl = 0; tl < 2; ++tl) {
    #pragma unroll
    for (int r = 0; r < 8; ++r) {
      const int m = tl * 16 + hi16 * 8 + r;
      out[((size_t)(g0 + m)) * DIM + u0] =
          accP[tl * 8 + r] * __builtin_amdgcn_rcpf(Zrun[tl * 8 + r]);
    }
  }
}

// ---------------- host entry ----------------
extern "C" void kernel_launch(void* const* d_in, const int* in_sizes, int n_in,
                              void* d_out, int out_size, void* d_ws, size_t ws_size,
                              hipStream_t stream) {
  (void)in_sizes; (void)n_in; (void)out_size; (void)ws_size;
  const float* x    = (const float*)d_in[0];
  // d_in[1] = batch (arange//SEQL) — pure reshape structure, not needed
  const float* Wih  = (const float*)d_in[2];
  const float* Whh  = (const float*)d_in[3];
  const float* bih  = (const float*)d_in[4];
  const float* bhh  = (const float*)d_in[5];
  const float* Watt = (const float*)d_in[6];
  float* out = (float*)d_out;

  unsigned short* wpack = (unsigned short*)d_ws;                       // 256 KB
  float* bias = (float*)((char*)d_ws + (size_t)GATES * KTOT * 2);      // 2 KB

  lstm_prep_kernel<<<512, 256, 0, stream>>>(Wih, Whh, bih, bhh, wpack, bias);

  (void)hipFuncSetAttribute((const void*)lstm_attn_kernel,
                            hipFuncAttributeMaxDynamicSharedMemorySize, SMEM_BYTES);
  lstm_attn_kernel<<<NBLK, NTHR, SMEM_BYTES, stream>>>(x, wpack, bias, Watt, out);
}
